// MultiHeadAttention_77369540870545
// MI455X (gfx1250) — compile-verified, matrix operand order checked by reading, not run on hardware
//
#include <hip/hip_runtime.h>
#include <hip/hip_bf16.h>

// ---------------------------------------------------------------------------
// Problem constants (match reference)
// ---------------------------------------------------------------------------
#define BATCH 2
#define TSEQ  2048
#define MDIM  2048
#define DHEAD 128
#define NHEAD 16
#define NROWS (BATCH * TSEQ)
#define QK_MULT 0.08838834764831845f   // (1/D)^0.5 applied to BOTH q and k
#define NEG_BIG (-3.0e38f)

typedef __attribute__((ext_vector_type(16))) __bf16 v16bf;
typedef __attribute__((ext_vector_type(8)))  __bf16 v8bf;
typedef __attribute__((ext_vector_type(4)))  __bf16 v4bf;
typedef __attribute__((ext_vector_type(8)))  float  v8f;
typedef __attribute__((ext_vector_type(4)))  float  v4f;

// ---------------------------------------------------------------------------
// WMMA helper (CDNA5 v_wmma_f32_16x16x32_bf16)
// ---------------------------------------------------------------------------
__device__ __forceinline__ v8f wmma_bf16(v16bf a, v16bf b, v8f c) {
    return __builtin_amdgcn_wmma_f32_16x16x32_bf16(
        false, a, false, b, (short)0, c, false, false);
}

// Row-contiguous 16x32 fragment load (works for A from row-major [m][k] and
// for B from *transposed* storage [n][k]).  Per lane: row = row0 + (lane&15),
// k-base = k0 + 8*(lane>=16); two contiguous 16-byte chunks at kb and kb+16.
__device__ __forceinline__ v16bf load_frag_row(const __bf16* __restrict__ base,
                                               int row0, int ld, int k0, int lane) {
    const int m  = row0 + (lane & 15);
    const int kb = k0 + ((lane >> 4) << 3);
    const __bf16* p = base + m * ld + kb;
    const v8bf c0 = *(const v8bf*)(p);
    const v8bf c1 = *(const v8bf*)(p + 16);
    return __builtin_shufflevector(c0, c1, 0, 1, 2, 3, 4, 5, 6, 7,
                                   8, 9, 10, 11, 12, 13, 14, 15);
}

// ---------------------------------------------------------------------------
// Kernel 1: fused QKV projection + centered-RMS norm + rotary + qk scaling.
// grid = (NROWS/128, NHEAD, 3)  block = 256 (8 waves)
// ---------------------------------------------------------------------------
__global__ __launch_bounds__(256) void qkv_proj_kernel(
    const float* __restrict__ x,  const float* __restrict__ wq,
    const float* __restrict__ wk, const float* __restrict__ wv,
    const float* __restrict__ gq, const float* __restrict__ bq,
    const float* __restrict__ gk, const float* __restrict__ bk,
    __bf16* __restrict__ q_ws, __bf16* __restrict__ k_ws,
    __bf16* __restrict__ v_ws)
{
    __shared__ __bf16 As[128 * 32];    // [m][k]  8 KB
    __shared__ __bf16 Bt[128 * 32];    // [n][k]  8 KB (transposed weight tile)
    __shared__ float  Cs[128 * 128];   // 64 KB

    const int tid  = threadIdx.x;
    const int lane = tid & 31;
    const int wid  = tid >> 5;
    const int row0 = blockIdx.x * 128;
    const int h    = blockIdx.y;
    const int col0 = h * DHEAD;
    const int mode = blockIdx.z;             // 0=q 1=k 2=v

    const float* w = (mode == 0) ? wq : (mode == 1) ? wk : wv;

    const int wm = (wid & 3) * 32;           // wave rows within tile
    const int wn = (wid >> 2) * 64;          // wave cols within tile

    v8f acc[2][4];
#pragma unroll
    for (int i = 0; i < 2; ++i)
#pragma unroll
        for (int j = 0; j < 4; ++j) acc[i][j] = (v8f){0.f,0.f,0.f,0.f,0.f,0.f,0.f,0.f};

    for (int kt = 0; kt < MDIM; kt += 32) {
        // stage A (128x32) fp32 -> bf16, float4 loads / v4bf stores
#pragma unroll
        for (int it = 0; it < 4; ++it) {
            const int idx = (tid + it * 256) * 4;        // 0..4095
            const int r = idx >> 5, c = idx & 31;
            const v4f f = *(const v4f*)&x[(size_t)(row0 + r) * MDIM + kt + c];
            *(v4bf*)&As[idx] = __builtin_convertvector(f, v4bf);
        }
        // stage B transposed: Bt[n][k] = w[(kt+k)*MDIM + col0 + n]
#pragma unroll
        for (int it = 0; it < 4; ++it) {
            const int item = tid + it * 256;             // 0..1023
            const int c = item & 127, r0 = (item >> 7) * 4;
            v4f f;
#pragma unroll
            for (int j = 0; j < 4; ++j)
                f[j] = w[(size_t)(kt + r0 + j) * MDIM + col0 + c];
            *(v4bf*)&Bt[c * 32 + r0] = __builtin_convertvector(f, v4bf);
        }
        __syncthreads();

        v16bf af[2], bf[4];
#pragma unroll
        for (int i = 0; i < 2; ++i) af[i] = load_frag_row(As, wm + i * 16, 32, 0, lane);
#pragma unroll
        for (int j = 0; j < 4; ++j) bf[j] = load_frag_row(Bt, wn + j * 16, 32, 0, lane);
#pragma unroll
        for (int i = 0; i < 2; ++i)
#pragma unroll
            for (int j = 0; j < 4; ++j) acc[i][j] = wmma_bf16(af[i], bf[j], acc[i][j]);
        __syncthreads();
    }

    // spill C tile to LDS
#pragma unroll
    for (int i = 0; i < 2; ++i)
#pragma unroll
        for (int j = 0; j < 4; ++j)
#pragma unroll
            for (int r = 0; r < 8; ++r) {
                const int rr = wm + i * 16 + r + ((lane & 16) ? 8 : 0);
                const int cc = wn + j * 16 + (lane & 15);
                Cs[rr * 128 + cc] = acc[i][j][r];
            }
    __syncthreads();

    // epilogue: per-row norm + rotary (q,k) or pass-through (v)
    if (tid < 128) {
        const int r  = tid;
        const int gr = row0 + r;
        const int bb = gr / TSEQ;
        const int t  = gr % TSEQ;
        if (mode == 2) {
            __bf16* dst = v_ws + ((size_t)(bb * NHEAD + h) * TSEQ + t) * DHEAD;
            for (int d = 0; d < DHEAD; d += 4) {
                const v4f f = *(const v4f*)&Cs[r * 128 + d];
                *(v4bf*)&dst[d] = __builtin_convertvector(f, v4bf);
            }
        } else {
            const float* g    = (mode == 0) ? gq : gk;
            const float* bvec = (mode == 0) ? bq : bk;
            float mu = 0.f;
            for (int d = 0; d < DHEAD; ++d) mu += Cs[r * 128 + d];
            mu *= (1.0f / DHEAD);
            float var = 0.f;
            for (int d = 0; d < DHEAD; ++d) {
                const float df = Cs[r * 128 + d] - mu;
                var += df * df;
            }
            var *= (1.0f / DHEAD);
            const float rinv = rsqrtf(var + 1e-6f);
            __bf16* dst = ((mode == 0) ? q_ws : k_ws) +
                          ((size_t)(bb * NHEAD + h) * TSEQ + t) * DHEAD;
            for (int d = 0; d < 64; ++d) {
                const float e = (Cs[r * 128 + d]      - mu) * rinv *
                                (1.f + g[h * DHEAD + d])      + bvec[h * DHEAD + d];
                const float o = (Cs[r * 128 + 64 + d] - mu) * rinv *
                                (1.f + g[h * DHEAD + 64 + d]) + bvec[h * DHEAD + 64 + d];
                const float freq = __powf(10000.f, -(float)d * (1.0f / 64.0f));
                const float rad  = (float)t * freq;
                const float cs = __cosf(rad);
                const float sn = __sinf(rad);
                dst[d]      = (__bf16)((e * cs - o * sn) * QK_MULT);
                dst[64 + d] = (__bf16)((e * sn + o * cs) * QK_MULT);
            }
        }
    }
}

// ---------------------------------------------------------------------------
// Kernel 2: causal flash attention.
// grid = (TSEQ/128, NHEAD, BATCH)  block = 256 (8 waves)
// Wave w owns S rows [w*16,+16) x 64 cols, O rows [w*16,+16) x 128 cols.
// ---------------------------------------------------------------------------
__global__ __launch_bounds__(256) void attn_kernel(
    const __bf16* __restrict__ q_ws, const __bf16* __restrict__ k_ws,
    const __bf16* __restrict__ v_ws, __bf16* __restrict__ o_ws)
{
    __shared__ __bf16 Ks[64 * 128];   // [key][d]  == [n][k] for S   16 KB
    __shared__ __bf16 Vt[128 * 64];   // [d][key]  == [n][k] for PV  16 KB
    __shared__ float  Ss[128 * 64];   // 32 KB
    __shared__ __bf16 Ps[128 * 64];   // [q][key]  A-side for PV     16 KB
    __shared__ float  mrow[128], lrow[128], srow[128];

    const int tid  = threadIdx.x;
    const int lane = tid & 31;
    const int wid  = tid >> 5;
    const int qt0  = blockIdx.x * 128;
    const int h    = blockIdx.y;
    const int bb   = blockIdx.z;

    const __bf16* qbase = q_ws + ((size_t)(bb * NHEAD + h) * TSEQ + qt0) * DHEAD;
    const __bf16* kbase = k_ws + ((size_t)(bb * NHEAD + h) * TSEQ) * DHEAD;
    const __bf16* vbase = v_ws + ((size_t)(bb * NHEAD + h) * TSEQ) * DHEAD;

    // preload Q fragments (this wave's 16 rows, 4 K-chunks) into VGPRs
    v16bf qf[4];
#pragma unroll
    for (int kc = 0; kc < 4; ++kc)
        qf[kc] = load_frag_row(qbase, wid * 16, DHEAD, kc * 32, lane);

    v8f oacc[8];
#pragma unroll
    for (int j = 0; j < 8; ++j) oacc[j] = (v8f){0.f,0.f,0.f,0.f,0.f,0.f,0.f,0.f};

    if (tid < 128) { mrow[tid] = NEG_BIG; lrow[tid] = 0.f; }
    __syncthreads();

    const int ntiles = 2 * blockIdx.x + 2;     // 64-key slabs up to diagonal
    for (int jt = 0; jt < ntiles; ++jt) {
        const int j64 = jt * 64;
        // stage K slab row-major (v8bf copies)
#pragma unroll
        for (int it = 0; it < 4; ++it) {
            const int idx = (tid + it * 256) * 8;        // 0..8191
            *(v8bf*)&Ks[idx] = *(const v8bf*)&kbase[(size_t)j64 * DHEAD + idx];
        }
        // stage V slab transposed: Vt[d][key]
#pragma unroll
        for (int it = 0; it < 8; ++it) {
            const int item = tid + it * 256;             // 0..2047
            const int d = item & 127, k0 = (item >> 7) * 4;
            v4bf vv;
#pragma unroll
            for (int j = 0; j < 4; ++j)
                vv[j] = vbase[(size_t)(j64 + k0 + j) * DHEAD + d];
            *(v4bf*)&Vt[d * 64 + k0] = vv;
        }
        __syncthreads();

        // S = Q * K^T  (wave: 16 rows x 64 cols -> 4 subtiles)
        v8f sacc[4];
#pragma unroll
        for (int j = 0; j < 4; ++j) sacc[j] = (v8f){0.f,0.f,0.f,0.f,0.f,0.f,0.f,0.f};
#pragma unroll
        for (int kc = 0; kc < 4; ++kc)
#pragma unroll
            for (int j = 0; j < 4; ++j) {
                const v16bf bt = load_frag_row(Ks, j * 16, DHEAD, kc * 32, lane);
                sacc[j] = wmma_bf16(qf[kc], bt, sacc[j]);
            }
#pragma unroll
        for (int j = 0; j < 4; ++j)
#pragma unroll
            for (int r = 0; r < 8; ++r) {
                const int rr = wid * 16 + r + ((lane & 16) ? 8 : 0);
                const int cc = j * 16 + (lane & 15);
                Ss[rr * 64 + cc] = sacc[j][r];
            }
        __syncthreads();

        // online softmax over this slab (one thread per query row)
        if (tid < 128) {
            const int r  = tid;
            const int qg = qt0 + r;
            const float mold = mrow[r];
            float mx = mold;
            for (int c = 0; c < 64; ++c)
                if (j64 + c <= qg) mx = fmaxf(mx, Ss[r * 64 + c]);
            const float sc = __expf(mold - mx);
            float sum = 0.f;
            for (int c = 0; c < 64; ++c) {
                const float p = (j64 + c <= qg) ? __expf(Ss[r * 64 + c] - mx) : 0.f;
                Ps[r * 64 + c] = (__bf16)p;
                sum += p;
            }
            lrow[r] = lrow[r] * sc + sum;
            mrow[r] = mx;
            srow[r] = sc;
        }
        __syncthreads();

        // rescale O accumulators, then O += P * V
#pragma unroll
        for (int j = 0; j < 8; ++j)
#pragma unroll
            for (int r = 0; r < 8; ++r) {
                const int rr = wid * 16 + r + ((lane & 16) ? 8 : 0);
                oacc[j][r] *= srow[rr];
            }
        v16bf pa[2];
#pragma unroll
        for (int kc = 0; kc < 2; ++kc)
            pa[kc] = load_frag_row(Ps, wid * 16, 64, kc * 32, lane);
#pragma unroll
        for (int j = 0; j < 8; ++j)
#pragma unroll
            for (int kc = 0; kc < 2; ++kc) {
                const v16bf bv = load_frag_row(Vt, j * 16, 64, kc * 32, lane);
                oacc[j] = wmma_bf16(pa[kc], bv, oacc[j]);
            }
        __syncthreads();   // before next slab overwrites Ks/Vt/Ps
    }

    // normalize and store O as [B, T, H*D] bf16
#pragma unroll
    for (int j = 0; j < 8; ++j)
#pragma unroll
        for (int r = 0; r < 8; ++r) {
            const int rr = wid * 16 + r + ((lane & 16) ? 8 : 0);
            const int cc = j * 16 + (lane & 15);
            const float val = oacc[j][r] / lrow[rr];
            const int t = qt0 + rr;
            o_ws[((size_t)(bb * TSEQ + t) * NHEAD + h) * DHEAD + cc] = (__bf16)val;
        }
}

// ---------------------------------------------------------------------------
// Kernel 3: output projection  out[4096,2048] = O[4096,2048] x Wo[2048,2048]
// grid = (NROWS/128, MDIM/128)  block = 256 (8 waves)
// ---------------------------------------------------------------------------
__global__ __launch_bounds__(256) void oproj_kernel(
    const __bf16* __restrict__ o_ws, const float* __restrict__ wo,
    float* __restrict__ out)
{
    __shared__ __bf16 As[128 * 32];   // [m][k]
    __shared__ __bf16 Bt[128 * 32];   // [n][k] transposed Wo tile

    const int tid  = threadIdx.x;
    const int lane = tid & 31;
    const int wid  = tid >> 5;
    const int row0 = blockIdx.x * 128;
    const int col0 = blockIdx.y * 128;

    const int wm = (wid & 3) * 32;
    const int wn = (wid >> 2) * 64;

    v8f acc[2][4];
#pragma unroll
    for (int i = 0; i < 2; ++i)
#pragma unroll
        for (int j = 0; j < 4; ++j) acc[i][j] = (v8f){0.f,0.f,0.f,0.f,0.f,0.f,0.f,0.f};

    for (int kt = 0; kt < MDIM; kt += 32) {
        // stage A (bf16 copy, v8bf)
#pragma unroll
        for (int it = 0; it < 2; ++it) {
            const int idx = (tid + it * 256) * 8;        // 0..4095
            const int r = idx >> 5, c = idx & 31;
            *(v8bf*)&As[idx] = *(const v8bf*)&o_ws[(size_t)(row0 + r) * MDIM + kt + c];
        }
        // stage Wo transposed: Bt[n][k]
#pragma unroll
        for (int it = 0; it < 4; ++it) {
            const int item = tid + it * 256;             // 0..1023
            const int c = item & 127, r0 = (item >> 7) * 4;
            v4f f;
#pragma unroll
            for (int j = 0; j < 4; ++j)
                f[j] = wo[(size_t)(kt + r0 + j) * MDIM + col0 + c];
            *(v4bf*)&Bt[c * 32 + r0] = __builtin_convertvector(f, v4bf);
        }
        __syncthreads();

        v16bf af[2], bf[4];
#pragma unroll
        for (int i = 0; i < 2; ++i) af[i] = load_frag_row(As, wm + i * 16, 32, 0, lane);
#pragma unroll
        for (int j = 0; j < 4; ++j) bf[j] = load_frag_row(Bt, wn + j * 16, 32, 0, lane);
#pragma unroll
        for (int i = 0; i < 2; ++i)
#pragma unroll
            for (int j = 0; j < 4; ++j) acc[i][j] = wmma_bf16(af[i], bf[j], acc[i][j]);
        __syncthreads();
    }

#pragma unroll
    for (int i = 0; i < 2; ++i)
#pragma unroll
        for (int j = 0; j < 4; ++j)
#pragma unroll
            for (int r = 0; r < 8; ++r) {
                const int rr = row0 + wm + i * 16 + r + ((lane & 16) ? 8 : 0);
                const int cc = col0 + wn + j * 16 + (lane & 15);
                out[(size_t)rr * MDIM + cc] = acc[i][j][r];
            }
}

// ---------------------------------------------------------------------------
// Host launcher
// ---------------------------------------------------------------------------
extern "C" void kernel_launch(void* const* d_in, const int* in_sizes, int n_in,
                              void* d_out, int out_size, void* d_ws, size_t ws_size,
                              hipStream_t stream) {
    const float* x  = (const float*)d_in[0];
    const float* wq = (const float*)d_in[1];
    const float* wk = (const float*)d_in[2];
    const float* wv = (const float*)d_in[3];
    const float* wo = (const float*)d_in[4];
    const float* gq = (const float*)d_in[5];
    const float* bq = (const float*)d_in[6];
    const float* gk = (const float*)d_in[7];
    const float* bk = (const float*)d_in[8];
    float* out = (float*)d_out;

    const size_t E = (size_t)BATCH * NHEAD * TSEQ * DHEAD;  // 8,388,608 elems
    __bf16* q_ws = (__bf16*)d_ws;
    __bf16* k_ws = q_ws + E;
    __bf16* v_ws = k_ws + E;
    __bf16* o_ws = v_ws + E;
    (void)in_sizes; (void)n_in; (void)out_size; (void)ws_size;

    qkv_proj_kernel<<<dim3(NROWS / 128, NHEAD, 3), 256, 0, stream>>>(
        x, wq, wk, wv, gq, bq, gk, bk, q_ws, k_ws, v_ws);

    attn_kernel<<<dim3(TSEQ / 128, NHEAD, BATCH), 256, 0, stream>>>(
        q_ws, k_ws, v_ws, o_ws);

    oproj_kernel<<<dim3(NROWS / 128, MDIM / 128), 256, 0, stream>>>(
        o_ws, wo, out);
}
